// LTFGW_semirelaxed_33681133535737
// MI455X (gfx1250) — matrix operand order repaired and checked by reading, not compiled, not run on hardware
//
#include <hip/hip_runtime.h>

// LTFGW semi-relaxed FGW, CDNA5 (gfx1250) wave32 + WMMA f32_16x16x32_f16.
// One wave per (node, template) pair. All tiny GEMMs (feature cross term,
// C1@T, (C1@T)@C2) run on the matrix pipes with f16 inputs / f32 accumulate.
// All WMMA operands are staged in LDS with layouts that make each lane's
// operand fragment contiguous (vector ds loads, no predicated global loads).

typedef __attribute__((ext_vector_type(16))) _Float16 v16h;
typedef __attribute__((ext_vector_type(8)))  _Float16 v8h;
typedef __attribute__((ext_vector_type(4)))  _Float16 v4h;
typedef __attribute__((ext_vector_type(8)))  float    v8f;

#define FDIM   128     // node feature dim
#define DEG    16      // fixed degree
#define MSUB   17      // subgraph size = DEG + 1
#define NT     10      // template nodes
#define NITER  10      // Frank-Wolfe iterations

struct __align__(64) Smem {
  _Float16 xsh [32][FDIM];  // A: subgraph features (rows 17..31 zero)      8192B
  _Float16 tfhT[16][FDIM];  // B: tfhT[col][k] = tfeat[t][col][k]           4096B
  _Float16 c1h [32][32];    // A: C1 padded                                 2048B
  _Float16 yh  [32][32];    // A: Y = C1@T padded                           2048B
  _Float16 thT [16][32];    // B: thT[col][k] = T[k][col]                   1024B
  _Float16 c2hT[16][32];    // B: c2hT[col][k] = C2[k][col]                 1024B
  float    mf  [32][16];    // feature cost M (f32)                         2048B
  float    wf  [32][16];    // W = C1@T@C2 (f32)                            2048B
  float    tf32[20][12];    // exact f32 copy of T (for q = colsum)          960B
  float    c2sq[10][12];    // C2 elementwise squared                        480B
  float    xs2[32];         // ||xs_r||^2 (rows >=17 zero)
  float    c1sp[32];        // (C1^2)@p per row
  float    qv[16];
  float    cc2[16];         // C2sq @ q
  float    f2c[16];         // template feature sq-norms (cols >=10 zero)
  float    red[32];
  int      sArr[20];
  int      maskA[20];
};

static __device__ __forceinline__ v16h loadA(const _Float16* lo, const _Float16* hi) {
  v8h l = *(const v8h*)lo;
  v8h h = *(const v8h*)hi;
  return __builtin_shufflevector(l, h, 0,1,2,3,4,5,6,7,8,9,10,11,12,13,14,15);
}

__global__ __launch_bounds__(32)
void ltfgw_srfgw_wmma_kernel(const float* __restrict__ x,
                             const int*   __restrict__ dst,
                             const float* __restrict__ tmpl,
                             const float* __restrict__ tfeat,
                             const float* __restrict__ alpha0,
                             float*       __restrict__ out,
                             int n, int Ntpl)
{
  __shared__ Smem sm;
  const int lane = threadIdx.x;
  const int node = blockIdx.x;
  const int t    = blockIdx.y;

  const float alpha = 1.0f / (1.0f + __expf(-alpha0[0]));
  const float p = 1.0f / (float)MSUB;

  // ---------------- stage 1: ids + zero-init padded tiles ----------------
  if (lane == 0)  sm.sArr[0] = node;
  if (lane < DEG) sm.sArr[1 + lane] = dst[node * DEG + lane];

  {
    const v8h z8 = {};
    if (lane >= MSUB) {
      #pragma unroll
      for (int c = 0; c < FDIM; c += 8) *(v8h*)&sm.xsh[lane][c] = z8;
    }
    if (lane >= NT && lane < 16) {
      #pragma unroll
      for (int c = 0; c < FDIM; c += 8) *(v8h*)&sm.tfhT[lane][c] = z8;
    }
    #pragma unroll
    for (int c = 0; c < 32; c += 8) *(v8h*)&sm.c1h[lane][c] = z8;
    #pragma unroll
    for (int c = 0; c < 32; c += 8) *(v8h*)&sm.yh[lane][c] = z8;
    if (lane < 16) {
      #pragma unroll
      for (int c = 0; c < 32; c += 8) *(v8h*)&sm.thT[lane][c] = z8;
      #pragma unroll
      for (int c = 0; c < 32; c += 8) *(v8h*)&sm.c2hT[lane][c] = z8;
      sm.f2c[lane] = 0.0f; sm.qv[lane] = 0.0f; sm.cc2[lane] = 0.0f;
    }
    sm.xs2[lane]  = 0.0f;
    sm.c1sp[lane] = 0.0f;
  }
  __syncthreads();

  // ---------------- stage 2: membership masks + vectorized gathers -------
  if (lane < MSUB) {
    const int sa = sm.sArr[lane];
    unsigned mask = 0u;
    for (int k = 0; k < DEG; ++k) {
      const int nb = dst[sa * DEG + k];
      #pragma unroll
      for (int b = 0; b < MSUB; ++b)
        if (nb == sm.sArr[b]) mask |= (1u << b);
    }
    sm.maskA[lane] = (int)mask;
  }
  // subgraph features: float4 global loads -> packed f16 LDS stores
  for (int idx = lane; idx < MSUB * (FDIM / 4); idx += 32) {
    const int r  = idx >> 5;            // FDIM/4 == 32
    const int c4 = idx & 31;
    const float4 v = *(const float4*)(x + (long)sm.sArr[r] * FDIM + c4 * 4);
    v4h h; h[0] = (_Float16)v.x; h[1] = (_Float16)v.y;
           h[2] = (_Float16)v.z; h[3] = (_Float16)v.w;
    *(v4h*)&sm.xsh[r][c4 * 4] = h;
  }
  // template features (B operand is the natural row-major layout)
  for (int idx = lane; idx < NT * (FDIM / 4); idx += 32) {
    const int r  = idx >> 5;
    const int c4 = idx & 31;
    const float4 v = *(const float4*)(tfeat + (long)(t * NT + r) * FDIM + c4 * 4);
    v4h h; h[0] = (_Float16)v.x; h[1] = (_Float16)v.y;
           h[2] = (_Float16)v.z; h[3] = (_Float16)v.w;
    *(v4h*)&sm.tfhT[r][c4 * 4] = h;
  }
  // C2 (transposed B layout) and C2^2
  for (int idx = lane; idx < NT * NT; idx += 32) {
    const int j = idx / NT, c = idx % NT;
    const float v = tmpl[t * NT * NT + j * NT + c];
    sm.c2hT[c][j] = (_Float16)v;           // [col][K]
    sm.c2sq[j][c] = v * v;                 // symmetric
  }
  __syncthreads();

  // ---------------- stage 3: C1, row norms, template norms ----------------
  if (lane < MSUB) {
    int rowsum = 0;
    #pragma unroll
    for (int b = 0; b < MSUB; ++b) {
      const int bit = ((sm.maskA[lane] >> b) | (sm.maskA[b] >> lane)) & 1;
      sm.c1h[lane][b] = (_Float16)(float)bit;
      rowsum += bit;
    }
    sm.c1sp[lane] = p * (float)rowsum;     // (C1^2)@p, C1 binary
    float s = 0.0f;
    for (int f = 0; f < FDIM; ++f) { const float v = (float)sm.xsh[lane][f]; s += v * v; }
    sm.xs2[lane] = s;
  }
  if (lane < NT) {
    float s = 0.0f;
    for (int f = 0; f < FDIM; ++f) { const float v = (float)sm.tfhT[lane][f]; s += v * v; }
    sm.f2c[lane] = s;
  }
  __syncthreads();

  // ---------------- stage 4: cross term via WMMA -> M ---------------------
  {
    const int row = lane & 15;
    const int hi  = (lane >> 4) & 1;
    const int col = lane & 15;
    #pragma unroll
    for (int mt = 0; mt < 2; ++mt) {
      v8f acc = {0.f,0.f,0.f,0.f,0.f,0.f,0.f,0.f};
      const int ar = row + 16 * mt;
      #pragma unroll
      for (int ks = 0; ks < 4; ++ks) {
        const int kA = ks * 32 + hi * 16;
        const v16h a = loadA(&sm.xsh[ar][kA], &sm.xsh[ar][kA + 8]);
        const int kB = ks * 32 + hi * 16;
        const v16h b = *(const v16h*)&sm.tfhT[col][kB];
        acc = __builtin_amdgcn_wmma_f32_16x16x32_f16(false, a, false, b,
                                                     (short)0, acc, false, false);
      }
      const int drb = hi * 8 + 16 * mt;
      #pragma unroll
      for (int v = 0; v < 8; ++v)
        sm.mf[drb + v][col] = sm.xs2[drb + v] + sm.f2c[col] - 2.0f * acc[v];
    }
  }

  // ---------------- stage 5: T init --------------------------------------
  float Trow[NT];
  if (lane < MSUB) {
    const float t0 = p * (1.0f / (float)NT);
    #pragma unroll
    for (int c = 0; c < NT; ++c) {
      Trow[c] = t0;
      sm.thT[c][lane]  = (_Float16)t0;     // [col][K]
      sm.tf32[lane][c] = t0;
    }
  }
  __syncthreads();

  // ---------------- stage 6: Frank-Wolfe loop -----------------------------
  for (int it = 0; it <= NITER; ++it) {
    // q = colsum(T), cc2 = C2sq @ q
    if (lane < NT) {
      float s = 0.0f;
      #pragma unroll
      for (int r = 0; r < MSUB; ++r) s += sm.tf32[r][lane];
      sm.qv[lane] = s;
    }
    __syncthreads();
    if (lane < NT) {
      float s = 0.0f;
      #pragma unroll
      for (int j = 0; j < NT; ++j) s += sm.c2sq[lane][j] * sm.qv[j];
      sm.cc2[lane] = s;
    }
    __syncthreads();

    const int row = lane & 15;
    const int hi  = (lane >> 4) & 1;
    const int col = lane & 15;

    // Y = C1 @ T  (K = 32 padded, one k-step, two M tiles)
    #pragma unroll
    for (int mt = 0; mt < 2; ++mt) {
      const int ar = row + 16 * mt;
      const int kA = hi * 16;
      const v16h a = loadA(&sm.c1h[ar][kA], &sm.c1h[ar][kA + 8]);
      const v16h b = *(const v16h*)&sm.thT[col][hi * 16];
      v8f acc = {0.f,0.f,0.f,0.f,0.f,0.f,0.f,0.f};
      acc = __builtin_amdgcn_wmma_f32_16x16x32_f16(false, a, false, b,
                                                   (short)0, acc, false, false);
      const int drb = hi * 8 + 16 * mt;
      #pragma unroll
      for (int v = 0; v < 8; ++v) sm.yh[drb + v][col] = (_Float16)acc[v];
    }
    __syncthreads();

    // W = Y @ C2
    #pragma unroll
    for (int mt = 0; mt < 2; ++mt) {
      const int ar = row + 16 * mt;
      const int kA = hi * 16;
      const v16h a = loadA(&sm.yh[ar][kA], &sm.yh[ar][kA + 8]);
      const v16h b = *(const v16h*)&sm.c2hT[col][hi * 16];
      v8f acc = {0.f,0.f,0.f,0.f,0.f,0.f,0.f,0.f};
      acc = __builtin_amdgcn_wmma_f32_16x16x32_f16(false, a, false, b,
                                                   (short)0, acc, false, false);
      const int drb = hi * 8 + 16 * mt;
      #pragma unroll
      for (int v = 0; v < 8; ++v) sm.wf[drb + v][col] = acc[v];
    }
    __syncthreads();

    if (it == NITER) break;   // final pass: W with the converged T, no update

    // gradient argmin per row + FW blend (lane r owns row r, exact f32)
    if (lane < MSUB) {
      float gmin = 3.0e38f; int jmin = 0;
      #pragma unroll
      for (int c = 0; c < NT; ++c) {
        const float g = (1.0f - alpha) * sm.mf[lane][c]
                      + 2.0f * alpha * (sm.c1sp[lane] + sm.cc2[c] - 2.0f * sm.wf[lane][c]);
        if (g < gmin) { gmin = g; jmin = c; }
      }
      const float gamma = 2.0f / ((float)it + 2.0f);
      #pragma unroll
      for (int c = 0; c < NT; ++c) Trow[c] *= (1.0f - gamma);
      Trow[jmin] += gamma * p;
      #pragma unroll
      for (int c = 0; c < NT; ++c) {
        sm.thT[c][lane]  = (_Float16)Trow[c];
        sm.tf32[lane][c] = Trow[c];
      }
    }
    __syncthreads();
  }

  // ---------------- stage 7: distance + reduction -------------------------
  float part = 0.0f;
  if (lane < MSUB) {
    #pragma unroll
    for (int c = 0; c < NT; ++c)
      part += Trow[c] * ((1.0f - alpha) * sm.mf[lane][c] - 2.0f * alpha * sm.wf[lane][c]);
    part += alpha * p * sm.c1sp[lane];            // p @ (C1^2) @ p contribution
  }
  if (lane < NT)
    part += alpha * sm.qv[lane] * sm.cc2[lane];   // q @ C2sq @ q contribution
  sm.red[lane] = part;
  __syncthreads();
  if (lane == 0) {
    float s = 0.0f;
    #pragma unroll
    for (int i = 0; i < 32; ++i) s += sm.red[i];
    out[(long)node * Ntpl + t] = s;
  }
}

extern "C" void kernel_launch(void* const* d_in, const int* in_sizes, int n_in,
                              void* d_out, int out_size, void* d_ws, size_t ws_size,
                              hipStream_t stream) {
  (void)n_in; (void)d_ws; (void)ws_size; (void)out_size;
  const float* x      = (const float*)d_in[0];
  const int*   ei     = (const int*)  d_in[1];
  const float* tmpl   = (const float*)d_in[2];
  const float* tfeat  = (const float*)d_in[3];
  const float* alpha0 = (const float*)d_in[4];
  float*       out    = (float*)d_out;

  const int n    = in_sizes[0] / FDIM;           // nodes
  const int E    = in_sizes[1] / 2;              // edges
  const int Ntpl = in_sizes[2] / (NT * NT);      // templates
  const int* dst = ei + E;                       // edge_index[1]

  dim3 grid((unsigned)n, (unsigned)Ntpl, 1);
  ltfgw_srfgw_wmma_kernel<<<grid, 32, 0, stream>>>(x, dst, tmpl, tfeat, alpha0,
                                                   out, n, Ntpl);
}
